// MultiHeadAttention_12987981103800
// MI455X (gfx1250) — matrix operand order repaired and checked by reading, not compiled, not run on hardware
//
#include <hip/hip_runtime.h>
#include <hip/hip_bf16.h>
#include <math.h>

typedef __bf16 bf16;
typedef __attribute__((ext_vector_type(16))) __bf16 v16bf;
typedef __attribute__((ext_vector_type(8)))  __bf16 v8bf;
typedef __attribute__((ext_vector_type(8)))  float  v8f;

#define B_    2
#define S_    2048
#define DM_   2048
#define NH_   16
#define HD_   128
#define ROWS_ (B_*S_)          // 4096 tokens total

// ---------------------------------------------------------------------------
// Fragment loaders.
// A-matrix 16-bit 16x32 layout used here (contiguous k): lane half hi=lane>>4 holds
//   elements 0..7  -> K = 16*hi + 0..7
//   elements 8..15 -> K = 16*hi + 8 + 0..7
// B-matrix 16-bit 32x16, same order here: lanes 0-15 hold K=0..15, lanes 16-31 hold K=16..31
// (contiguous), so a B fragment is one 32-byte load.
// ---------------------------------------------------------------------------
__device__ __forceinline__ v16bf load_a_frag(const bf16* base, int hi) {
  v8bf lo = *(const v8bf*)(base + 16*hi);
  v8bf hh = *(const v8bf*)(base + 16*hi + 8);
  v16bf a;
#pragma unroll
  for (int e = 0; e < 8; ++e) { a[e] = lo[e]; a[8+e] = hh[e]; }
  return a;
}

__device__ __forceinline__ v8f wmma_bf16(v16bf a, v16bf b, v8f c) {
  return __builtin_amdgcn_wmma_f32_16x16x32_bf16(false, a, false, b,
                                                 (short)0, c, false, false);
}

// ---------------------------------------------------------------------------
// Two-stage ping-pong pipelined 16x64 wave-tile GEMM core over K = DM_.
// Stage = 1 A-frag + 4 B-frags (40 VGPRs); consume fX then refill fX, so the
// allocator coalesces loads into the same registers (no rotation copies) and
// total pressure (acc 32 + 2x40) stays well inside the low-256 VGPR window.
// ---------------------------------------------------------------------------
__device__ __forceinline__ void gemm_k_loop(const bf16* arow, const bf16* bbase,
                                            v8f acc[4], int hi) {
  v16bf aA = load_a_frag(arow, hi);
  v16bf bA[4];
#pragma unroll
  for (int j = 0; j < 4; ++j)
    bA[j] = *(const v16bf*)(bbase + (size_t)j * 16 * DM_);
  v16bf aB = load_a_frag(arow + 32, hi);
  v16bf bB[4];
#pragma unroll
  for (int j = 0; j < 4; ++j)
    bB[j] = *(const v16bf*)(bbase + (size_t)j * 16 * DM_ + 32);

  for (int k0 = 0; k0 < DM_ - 64; k0 += 64) {
    __builtin_prefetch(arow + k0 + 576, 0, 3);
    // consume stage A (loads issued one full stage ago)
#pragma unroll
    for (int j = 0; j < 4; ++j) acc[j] = wmma_bf16(aA, bA[j], acc[j]);
    // refill stage A
    aA = load_a_frag(arow + k0 + 64, hi);
#pragma unroll
    for (int j = 0; j < 4; ++j)
      bA[j] = *(const v16bf*)(bbase + (size_t)j * 16 * DM_ + k0 + 64);
    // consume stage B
#pragma unroll
    for (int j = 0; j < 4; ++j) acc[j] = wmma_bf16(aB, bB[j], acc[j]);
    // refill stage B
    aB = load_a_frag(arow + k0 + 96, hi);
#pragma unroll
    for (int j = 0; j < 4; ++j)
      bB[j] = *(const v16bf*)(bbase + (size_t)j * 16 * DM_ + k0 + 96);
  }
#pragma unroll
  for (int j = 0; j < 4; ++j) acc[j] = wmma_bf16(aA, bA[j], acc[j]);   // k = DM-64
#pragma unroll
  for (int j = 0; j < 4; ++j) acc[j] = wmma_bf16(aB, bB[j], acc[j]);   // k = DM-32
}

// ---------------------------------------------------------------------------
// Elementwise f32 -> bf16 conversion
// ---------------------------------------------------------------------------
__global__ __launch_bounds__(256) void k_convert_bf16(const float* __restrict__ in,
                                                      bf16* __restrict__ out, int n) {
  int i = blockIdx.x * blockDim.x + threadIdx.x;
  if (i < n) out[i] = (bf16)in[i];
}

// ---------------------------------------------------------------------------
// Transpose + convert: out[n*rows + k] = (bf16) in[k*cols + n]
// 32x32 LDS tile, block (32,8)
// ---------------------------------------------------------------------------
__global__ __launch_bounds__(256) void k_transpose_bf16(const float* __restrict__ in,
                                                        bf16* __restrict__ out,
                                                        int rows, int cols) {
  __shared__ float tile[32][33];
  int c0 = blockIdx.x * 32, r0 = blockIdx.y * 32;
  int tx = threadIdx.x, ty = threadIdx.y;
#pragma unroll
  for (int i = 0; i < 32; i += 8)
    tile[ty + i][tx] = in[(size_t)(r0 + ty + i) * cols + (c0 + tx)];
  __syncthreads();
#pragma unroll
  for (int i = 0; i < 32; i += 8)
    out[(size_t)(c0 + ty + i) * rows + (r0 + tx)] = (bf16)tile[tx][ty + i];
}

// ---------------------------------------------------------------------------
// QKV projection GEMM (bf16 WMMA, f32 accum) + bias + RoPE epilogue.
// Wave: 16 rows x 64 cols; block 256 = 8 waves -> 128x64 tile; z selects Q/K/V.
// Q,K written [b,h,s,d] bf16 (post-RoPE); V written transposed [b,h,d,s] bf16.
// ---------------------------------------------------------------------------
__global__ __launch_bounds__(256) void k_qkv_gemm_rope(
    const bf16* __restrict__ Xb,
    const bf16* __restrict__ WqT, const bf16* __restrict__ WkT, const bf16* __restrict__ WvT,
    const float* __restrict__ bq, const float* __restrict__ bk, const float* __restrict__ bv,
    bf16* __restrict__ Qb, bf16* __restrict__ Kb, bf16* __restrict__ VTb) {
  const int mode = blockIdx.z;                       // 0=Q 1=K 2=V
  const bf16*  WT   = (mode == 0) ? WqT : (mode == 1) ? WkT : WvT;
  const float* bias = (mode == 0) ? bq  : (mode == 1) ? bk  : bv;

  const int lane = threadIdx.x & 31;
  const int wid  = threadIdx.x >> 5;
  const int lr   = lane & 15;
  const int hi   = lane >> 4;

  const int n0 = blockIdx.x * 64;
  const int m0 = blockIdx.y * 128 + wid * 16;

  v8f acc[4];
#pragma unroll
  for (int j = 0; j < 4; ++j)
#pragma unroll
    for (int r = 0; r < 8; ++r) acc[j][r] = 0.0f;

  const bf16* arow  = Xb + (size_t)(m0 + lr) * DM_;
  const bf16* bbase = WT + (size_t)(n0 + lr) * DM_ + 16*hi;
  gemm_k_loop(arow, bbase, acc, hi);

  // Epilogue: C layout -> element r, lane: row m0 + r + 8*hi, col n0+j*16+lr
#pragma unroll
  for (int j = 0; j < 4; ++j) {
    const int n = n0 + j*16 + lr;
    const int h = n / HD_, d = n % HD_;
    const float bsv = bias[n];
    if (mode < 2) {
      bf16* OUT = (mode == 0) ? Qb : Kb;
      const float freq = __powf(10000.0f, -(float)(2*(d >> 1)) / (float)HD_);
      const float sgn  = (d & 1) ? 1.0f : -1.0f;     // even d: -x2*sin ; odd d: +x1*sin
#pragma unroll
      for (int r = 0; r < 8; ++r) {
        const int row = m0 + r + 8*hi;
        const int bi = row / S_, s = row % S_;
        float val = acc[j][r] + bsv;
        float partner = __shfl_xor(val, 1, 32);      // adjacent feature lives in lane^1
        float sn, cs;
        __sincosf((float)s * freq, &sn, &cs);
        float o = val * cs + sgn * partner * sn;
        OUT[((size_t)(bi*NH_ + h)*S_ + s) * HD_ + d] = (bf16)o;
      }
    } else {
#pragma unroll
      for (int r = 0; r < 8; ++r) {
        const int row = m0 + r + 8*hi;
        const int bi = row / S_, s = row % S_;
        float val = acc[j][r] + bsv;
        VTb[((size_t)(bi*NH_ + h)*HD_ + d) * S_ + s] = (bf16)val;    // V transposed
      }
    }
  }
}

// ---------------------------------------------------------------------------
// Causal flash attention: one wave per (b,h,16-query tile); 32 keys per step.
// QK^T and P.V both via bf16 WMMA; online softmax in f32.
// V^T fragments preloaded before the softmax section to hide global latency.
// ---------------------------------------------------------------------------
__global__ __launch_bounds__(128) void k_flash_attn(
    const bf16* __restrict__ Qb, const bf16* __restrict__ Kb,
    const bf16* __restrict__ VTb, bf16* __restrict__ Ob) {
  __shared__ bf16 plds[4][16 * 32];                  // wave-private P tiles

  const int lane  = threadIdx.x & 31;
  const int wslot = threadIdx.x >> 5;
  const int lr = lane & 15, hi = lane >> 4;

  const int w  = blockIdx.x * 4 + wslot;
  const int bh = w >> 7;                             // 128 q-tiles per head
  const int q0 = (w & 127) * 16;

  const bf16* Qh = Qb  + ((size_t)bh * S_ + q0) * HD_;
  const bf16* Kh = Kb  + (size_t)bh * S_ * HD_;
  const bf16* Vh = VTb + (size_t)bh * HD_ * S_;

  v16bf qf[4];
#pragma unroll
  for (int t = 0; t < 4; ++t) qf[t] = load_a_frag(Qh + lr*HD_ + 32*t, hi);

  v8f O[8];
  float mi[8], li[8];
#pragma unroll
  for (int dt = 0; dt < 8; ++dt)
#pragma unroll
    for (int r = 0; r < 8; ++r) O[dt][r] = 0.0f;
#pragma unroll
  for (int r = 0; r < 8; ++r) { mi[r] = -1e30f; li[r] = 0.0f; }

  const float scale = 0.08838834764831845f;          // 1/sqrt(128)
  const int kmax = q0 + 16;

  for (int k0 = 0; k0 < kmax; k0 += 32) {
    if (k0 + 32 < kmax)                              // prefetch next key tile (32 rows)
      __builtin_prefetch(Kh + (size_t)(k0 + 32 + lane) * HD_, 0, 3);

    // ---- scores: two 16x16 sub-tiles over 32 keys ----
    v8f sc[2];
#pragma unroll
    for (int s2 = 0; s2 < 2; ++s2) {
      int kc = k0 + 16*s2 + lr; if (kc > S_ - 1) kc = S_ - 1;
      const bf16* kptr = Kh + (size_t)kc * HD_;
      v8f a;
#pragma unroll
      for (int r = 0; r < 8; ++r) a[r] = 0.0f;
#pragma unroll
      for (int t = 0; t < 4; ++t) {
        v16bf bb = *(const v16bf*)(kptr + 32*t + 16*hi);
        a = wmma_bf16(qf[t], bb, a);
      }
#pragma unroll
      for (int r = 0; r < 8; ++r) {
        const int q  = q0 + r + 8*hi;
        const int kk = k0 + 16*s2 + lr;
        sc[s2][r] = (kk > q) ? -1e30f : a[r] * scale;
      }
    }

    // ---- preload V^T fragments now; they complete under the softmax math ----
    int kst = k0 + 16*hi; if (kst > S_ - 16) kst = S_ - 16;
    v16bf vb[8];
#pragma unroll
    for (int dt = 0; dt < 8; ++dt)
      vb[dt] = *(const v16bf*)(Vh + (size_t)(16*dt + lr) * S_ + kst);

    // ---- online softmax: row reductions across 16-lane half groups ----
    float mnew[8], alpha[8];
#pragma unroll
    for (int r = 0; r < 8; ++r) {
      float m = fmaxf(sc[0][r], sc[1][r]);
      m = fmaxf(m, __shfl_xor(m, 1, 32));
      m = fmaxf(m, __shfl_xor(m, 2, 32));
      m = fmaxf(m, __shfl_xor(m, 4, 32));
      m = fmaxf(m, __shfl_xor(m, 8, 32));
      mnew[r]  = fmaxf(mi[r], m);
      alpha[r] = __expf(mi[r] - mnew[r]);
      mi[r]    = mnew[r];
    }
#pragma unroll
    for (int s2 = 0; s2 < 2; ++s2)
#pragma unroll
      for (int r = 0; r < 8; ++r) sc[s2][r] = __expf(sc[s2][r] - mnew[r]);
#pragma unroll
    for (int r = 0; r < 8; ++r) {
      float rs = sc[0][r] + sc[1][r];
      rs += __shfl_xor(rs, 1, 32);
      rs += __shfl_xor(rs, 2, 32);
      rs += __shfl_xor(rs, 4, 32);
      rs += __shfl_xor(rs, 8, 32);
      li[r] = li[r] * alpha[r] + rs;
    }
#pragma unroll
    for (int dt = 0; dt < 8; ++dt)
#pragma unroll
      for (int r = 0; r < 8; ++r) O[dt][r] *= alpha[r];

    // ---- relayout exp(P): C layout -> A fragment via wave-private LDS ----
#pragma unroll
    for (int s2 = 0; s2 < 2; ++s2)
#pragma unroll
      for (int r = 0; r < 8; ++r)
        plds[wslot][(r + 8*hi) * 32 + 16*s2 + lr] = (bf16)sc[s2][r];
    // same-wave LDS ops are in-order; no cross-wave sharing -> no barrier
    v8bf plo = *(const v8bf*)&plds[wslot][lr*32 + 16*hi];
    v8bf phh = *(const v8bf*)&plds[wslot][lr*32 + 16*hi + 8];
    v16bf pa;
#pragma unroll
    for (int e = 0; e < 8; ++e) { pa[e] = plo[e]; pa[8+e] = phh[e]; }

    // ---- O += P @ V : 8 head-dim tiles (V fragments already in registers) ----
#pragma unroll
    for (int dt = 0; dt < 8; ++dt)
      O[dt] = wmma_bf16(pa, vb[dt], O[dt]);
  }

  // ---- epilogue: normalize and write [b, s, h*HD + d] as bf16 ----
  const int b = bh >> 4, h = bh & 15;
#pragma unroll
  for (int r = 0; r < 8; ++r) {
    const float inv = 1.0f / li[r];
    const int row = b * S_ + q0 + r + 8*hi;
#pragma unroll
    for (int dt = 0; dt < 8; ++dt)
      Ob[(size_t)row * DM_ + h*HD_ + 16*dt + lr] = (bf16)(O[dt][r] * inv);
  }
}

// ---------------------------------------------------------------------------
// Output projection: attn[ROWS][DM] bf16 @ Wo^T bf16 + bo -> f32 out
// Same pipelined 16x64 wave tile.
// ---------------------------------------------------------------------------
__global__ __launch_bounds__(256) void k_out_gemm(
    const bf16* __restrict__ Ab, const bf16* __restrict__ WoT,
    const float* __restrict__ bo, float* __restrict__ out) {
  const int lane = threadIdx.x & 31;
  const int wid  = threadIdx.x >> 5;
  const int lr = lane & 15, hi = lane >> 4;

  const int n0 = blockIdx.x * 64;
  const int m0 = blockIdx.y * 128 + wid * 16;

  v8f acc[4];
#pragma unroll
  for (int j = 0; j < 4; ++j)
#pragma unroll
    for (int r = 0; r < 8; ++r) acc[j][r] = 0.0f;

  const bf16* arow  = Ab + (size_t)(m0 + lr) * DM_;
  const bf16* bbase = WoT + (size_t)(n0 + lr) * DM_ + 16*hi;
  gemm_k_loop(arow, bbase, acc, hi);

#pragma unroll
  for (int j = 0; j < 4; ++j) {
    const int n = n0 + j*16 + lr;
    const float bsv = bo[n];
#pragma unroll
    for (int r = 0; r < 8; ++r)
      out[(size_t)(m0 + r + 8*hi) * DM_ + n] = acc[j][r] + bsv;
  }
}

// ---------------------------------------------------------------------------
// Launch
// ---------------------------------------------------------------------------
extern "C" void kernel_launch(void* const* d_in, const int* in_sizes, int n_in,
                              void* d_out, int out_size, void* d_ws, size_t ws_size,
                              hipStream_t stream) {
  const float* hs = (const float*)d_in[0];
  const float* Wq = (const float*)d_in[1];
  const float* bq = (const float*)d_in[2];
  const float* Wk = (const float*)d_in[3];
  const float* bk = (const float*)d_in[4];
  const float* Wv = (const float*)d_in[5];
  const float* bv = (const float*)d_in[6];
  const float* Wo = (const float*)d_in[7];
  const float* bo = (const float*)d_in[8];
  float* out = (float*)d_out;

  const size_t MB = 1ull << 20;
  char* ws = (char*)d_ws;
  bf16* hsb = (bf16*)(ws +   0*MB);   // 16 MB  hidden bf16
  bf16* WqT = (bf16*)(ws +  16*MB);   //  8 MB
  bf16* WkT = (bf16*)(ws +  24*MB);
  bf16* WvT = (bf16*)(ws +  32*MB);
  bf16* WoT = (bf16*)(ws +  40*MB);
  bf16* Qb  = (bf16*)(ws +  48*MB);   // 16 MB  [b,h,s,d]
  bf16* Kb  = (bf16*)(ws +  64*MB);   // 16 MB  [b,h,s,d]
  bf16* VTb = (bf16*)(ws +  80*MB);   // 16 MB  [b,h,d,s]
  bf16* Ab  = (bf16*)(ws +  96*MB);   // 16 MB  attn out [b,s,dm]

  const int nhs = ROWS_ * DM_;
  k_convert_bf16<<<(nhs + 255)/256, 256, 0, stream>>>(hs, hsb, nhs);

  dim3 tb(32, 8), tg(DM_/32, DM_/32);
  k_transpose_bf16<<<tg, tb, 0, stream>>>(Wq, WqT, DM_, DM_);
  k_transpose_bf16<<<tg, tb, 0, stream>>>(Wk, WkT, DM_, DM_);
  k_transpose_bf16<<<tg, tb, 0, stream>>>(Wv, WvT, DM_, DM_);
  k_transpose_bf16<<<tg, tb, 0, stream>>>(Wo, WoT, DM_, DM_);

  dim3 gg(DM_/64, ROWS_/128, 3);
  k_qkv_gemm_rope<<<gg, 256, 0, stream>>>(hsb, WqT, WkT, WvT, bq, bk, bv, Qb, Kb, VTb);

  const int nwaves = B_ * NH_ * (S_/16);             // 4096 waves
  k_flash_attn<<<nwaves/4, 128, 0, stream>>>(Qb, Kb, VTb, Ab);

  dim3 go(DM_/64, ROWS_/128, 1);
  k_out_gemm<<<go, 256, 0, stream>>>(Ab, WoT, bo, out);
}
